// Distiller_51960514347089
// MI455X (gfx1250) — matrix operand order repaired and checked by reading, not compiled, not run on hardware
//
#include <hip/hip_runtime.h>
#include <math.h>

typedef __attribute__((ext_vector_type(16))) _Float16 v16h;
typedef __attribute__((ext_vector_type(8)))  _Float16 v8h;
typedef __attribute__((ext_vector_type(8)))  float    v8f;

#define N_F3   589824   // 4*64*48*48
#define N_OUT  1376256  // 4*21*128*128
#define SA_M   2304
#define SA_NT  144      // 2304/16
#define PI_BLOCKS 256

// ---------------------------------------------------------------- copy s_out
__global__ void copy_kernel(const float* __restrict__ src, float* __restrict__ dst, int n) {
    int i = blockIdx.x * blockDim.x + threadIdx.x;
    if (i < n) dst[i] = src[i];
}

// ---------------------------------------------------------------- f32 -> f16
__global__ void f32_to_f16_kernel(const float* __restrict__ src, _Float16* __restrict__ dst, int n) {
    int i = blockIdx.x * blockDim.x + threadIdx.x;
    if (i < n) dst[i] = (_Float16)src[i];
}

// ---------------------------------------------------------------- dual conv3x3 (f16 out)
// x: (4,64,48,48), W: (64,64,3,3); raw-view rows (m*64..m*64+63) stay contiguous
__global__ void conv_dual_kernel(const float* __restrict__ x,
                                 const float* __restrict__ WB, const float* __restrict__ bB,
                                 const float* __restrict__ WC, const float* __restrict__ bC,
                                 _Float16* __restrict__ Ab, _Float16* __restrict__ Ac) {
    int idx = blockIdx.x * blockDim.x + threadIdx.x;
    if (idx >= N_F3) return;
    int b  = idx / (64 * 2304);
    int co = (idx / 2304) % 64;
    int y  = (idx % 2304) / 48;
    int xw = idx % 48;
    float accB = bB[co], accC = bC[co];
    for (int ci = 0; ci < 64; ++ci) {
        const float* xin = x + ((b * 64 + ci) * 48) * 48;
        const float* wB = WB + (co * 64 + ci) * 9;
        const float* wC = WC + (co * 64 + ci) * 9;
        #pragma unroll
        for (int ky = 0; ky < 3; ++ky) {
            int iy = y + ky - 1;
            if (iy < 0 || iy >= 48) continue;
            #pragma unroll
            for (int kx = 0; kx < 3; ++kx) {
                int ix = xw + kx - 1;
                if (ix < 0 || ix >= 48) continue;
                float v = xin[iy * 48 + ix];
                accB = fmaf(v, wB[ky * 3 + kx], accB);
                accC = fmaf(v, wC[ky * 3 + kx], accC);
            }
        }
    }
    Ab[idx] = (_Float16)accB;
    Ac[idx] = (_Float16)accC;
}

// ---------------------------------------------------------------- SA strip (WMMA)
static __device__ inline v8f wmma_f16(v16h a, v16h b, v8f c) {
    return __builtin_amdgcn_wmma_f32_16x16x32_f16(false, a, false, b, (short)0, c,
                                                  false, false);
}

// A-matrix fragment (16x32 f16): lane row = lane&15, kb = (lane>=16)*8,
// halves h -> K = 2*kb+h (contiguous, matches B). Two contiguous v8h segments.
static __device__ inline v16h load_afrag(const _Float16* __restrict__ row, int koff, int kb) {
    v8h lo = *(const v8h*)(row + koff + 2 * kb);
    v8h hi = *(const v8h*)(row + koff + 2 * kb + 8);
    v16h a;
    #pragma unroll
    for (int h = 0; h < 8; ++h) { a[h] = lo[h]; a[h + 8] = hi[h]; }
    return a;
}

__global__ __launch_bounds__(256)
void sa_strip_kernel(const _Float16* __restrict__ Th, const _Float16* __restrict__ Abh,
                     const _Float16* __restrict__ Ach, float* __restrict__ saPart) {
    int blk = blockIdx.x;                 // 0..575
    int b   = blk / SA_NT;
    int m0  = (blk % SA_NT) * 16;
    int tid  = threadIdx.x;
    int lane = tid & 31;
    int wave = tid >> 5;                  // 8 waves, each owns 18 n-tiles

    __shared__ float stX[8][16][2];       // per-wave merged (max,sum) for X rows
    __shared__ float stS[8][16][2];
    __shared__ float fin[16][3];          // lseX, mS, 1/ZS per row
    __shared__ float wred[8];

    const _Float16* Tb  = Th  + (size_t)b * SA_M * 64;
    const _Float16* Abb = Abh + (size_t)b * SA_M * 64;
    const _Float16* Acb = Ach + (size_t)b * SA_M * 64;

    int rA  = lane & 15;
    int kb  = (lane >> 4) * 8;            // A-frag K sub-base
    int nC  = lane & 15;                  // B-frag N / C column
    int kbB = (lane >> 4) * 16;           // B-frag K base
    int rowBase = (lane < 16) ? 0 : 8;
    const float invs = 1.0f / 48.0f;      // 1/sqrt(M)

    // A fragments: loaded once, straight from global f16
    const _Float16* rowT = Tb  + (size_t)(m0 + rA) * 64;
    const _Float16* rowB = Abb + (size_t)(m0 + rA) * 64;
    v16h aT0 = load_afrag(rowT, 0,  kb);
    v16h aT1 = load_afrag(rowT, 32, kb);
    v16h aB0 = load_afrag(rowB, 0,  kb);
    v16h aB1 = load_afrag(rowB, 32, kb);

    // lane-local online stats (no shuffles in hot loop)
    float mXl[8], sXl[8], mSl[8], sSl[8];
    #pragma unroll
    for (int r = 0; r < 8; ++r) { mXl[r] = -1e30f; sXl[r] = 0.f; mSl[r] = -1e30f; sSl[r] = 0.f; }

    // ---- pass 0: stats ----
    for (int t = 0; t < SA_NT / 8; ++t) {
        int n0 = (wave + t * 8) * 16;
        const _Float16* pT = Tb  + (size_t)(n0 + nC) * 64 + kbB;
        const _Float16* pC = Acb + (size_t)(n0 + nC) * 64 + kbB;
        if (t + 1 < SA_NT / 8) {
            __builtin_prefetch(pT + 8 * 16 * 64, 0, 0);
            __builtin_prefetch(pC + 8 * 16 * 64, 0, 0);
        }
        v16h bT0 = *(const v16h*)(pT);
        v16h bT1 = *(const v16h*)(pT + 32);
        v16h bC0 = *(const v16h*)(pC);
        v16h bC1 = *(const v16h*)(pC + 32);
        v8f accX = {}; v8f accS = {};
        accX = wmma_f16(aT0, bT0, accX);
        accX = wmma_f16(aT1, bT1, accX);
        accS = wmma_f16(aB0, bC0, accS);
        accS = wmma_f16(aB1, bC1, accS);
        int gn = n0 + nC;
        #pragma unroll
        for (int r = 0; r < 8; ++r) {
            float x = accX[r] * invs;
            float s = accS[r] * invs;
            if (m0 + rowBase + r == gn) { x = 0.f; s = 0.f; }   // * (1 - eye)
            float nm = fmaxf(mXl[r], x);
            sXl[r] = sXl[r] * __expf(mXl[r] - nm) + __expf(x - nm);
            mXl[r] = nm;
            float nm2 = fmaxf(mSl[r], s);
            sSl[r] = sSl[r] * __expf(mSl[r] - nm2) + __expf(s - nm2);
            mSl[r] = nm2;
        }
    }

    // merge stats across the 16 lanes of each half (log-sum-exp combine)
    #pragma unroll
    for (int r = 0; r < 8; ++r) {
        #pragma unroll
        for (int o = 1; o < 16; o <<= 1) {
            float om = __shfl_xor(mXl[r], o), os = __shfl_xor(sXl[r], o);
            float nm = fmaxf(mXl[r], om);
            sXl[r] = sXl[r] * __expf(mXl[r] - nm) + os * __expf(om - nm);
            mXl[r] = nm;
            float om2 = __shfl_xor(mSl[r], o), os2 = __shfl_xor(sSl[r], o);
            float nm2 = fmaxf(mSl[r], om2);
            sSl[r] = sSl[r] * __expf(mSl[r] - nm2) + os2 * __expf(om2 - nm2);
            mSl[r] = nm2;
        }
    }
    if (lane == 0 || lane == 16) {
        #pragma unroll
        for (int r = 0; r < 8; ++r) {
            stX[wave][rowBase + r][0] = mXl[r];
            stX[wave][rowBase + r][1] = sXl[r];
            stS[wave][rowBase + r][0] = mSl[r];
            stS[wave][rowBase + r][1] = sSl[r];
        }
    }
    __syncthreads();
    if (tid < 16) {
        float m = -1e30f, s = 0.f, m2 = -1e30f, s2 = 0.f;
        for (int w = 0; w < 8; ++w) {
            float om = stX[w][tid][0], os = stX[w][tid][1];
            float nm = fmaxf(m, om);
            s = s * __expf(m - nm) + os * __expf(om - nm);
            m = nm;
            float om2 = stS[w][tid][0], os2 = stS[w][tid][1];
            float nm2 = fmaxf(m2, om2);
            s2 = s2 * __expf(m2 - nm2) + os2 * __expf(om2 - nm2);
            m2 = nm2;
        }
        fin[tid][0] = m + logf(s);    // lseX
        fin[tid][1] = m2;             // max S
        fin[tid][2] = 1.0f / s2;      // 1/ZS
    }
    __syncthreads();
    float lse[8], mSf[8], iZ[8];
    #pragma unroll
    for (int r = 0; r < 8; ++r) {
        lse[r] = fin[rowBase + r][0];
        mSf[r] = fin[rowBase + r][1];
        iZ[r]  = fin[rowBase + r][2];
    }

    // ---- pass 1: recompute tiles, accumulate sum((log_softmax X - softmax S)^2) ----
    float acc2 = 0.f;
    for (int t = 0; t < SA_NT / 8; ++t) {
        int n0 = (wave + t * 8) * 16;
        const _Float16* pT = Tb  + (size_t)(n0 + nC) * 64 + kbB;
        const _Float16* pC = Acb + (size_t)(n0 + nC) * 64 + kbB;
        v16h bT0 = *(const v16h*)(pT);
        v16h bT1 = *(const v16h*)(pT + 32);
        v16h bC0 = *(const v16h*)(pC);
        v16h bC1 = *(const v16h*)(pC + 32);
        v8f accX = {}; v8f accS = {};
        accX = wmma_f16(aT0, bT0, accX);
        accX = wmma_f16(aT1, bT1, accX);
        accS = wmma_f16(aB0, bC0, accS);
        accS = wmma_f16(aB1, bC1, accS);
        int gn = n0 + nC;
        #pragma unroll
        for (int r = 0; r < 8; ++r) {
            float x = accX[r] * invs;
            float s = accS[r] * invs;
            if (m0 + rowBase + r == gn) { x = 0.f; s = 0.f; }
            float xl = x - lse[r];
            float sm = __expf(s - mSf[r]) * iZ[r];
            float d = xl - sm;
            acc2 = fmaf(d, d, acc2);
        }
    }
    #pragma unroll
    for (int o = 1; o < 32; o <<= 1) acc2 += __shfl_xor(acc2, o);
    if (lane == 0) wred[wave] = acc2;
    __syncthreads();
    if (tid == 0) {
        float s = 0.f;
        for (int w = 0; w < 8; ++w) s += wred[w];
        saPart[blk] = s;
    }
}

// ---------------------------------------------------------------- block reduce
static __device__ inline void block_reduce_store(float v, float* dst) {
    __shared__ float red[256];
    int t = threadIdx.x;
    red[t] = v;
    __syncthreads();
    for (int s = 128; s > 0; s >>= 1) {
        if (t < s) red[t] += red[t + s];
        __syncthreads();
    }
    if (t == 0) *dst = red[0];
}

// ---------------------------------------------------------------- pi_loss partials
__global__ void pi_kernel(const float* __restrict__ t_out, const float* __restrict__ s_out,
                          float* __restrict__ piPart) {
    int p = blockIdx.x * blockDim.x + threadIdx.x;   // pixel 0..65535
    int b  = p >> 14;
    int hw = p & 16383;
    const float* tb = t_out + (size_t)b * 21 * 16384 + hw;
    const float* sb = s_out + (size_t)b * 21 * 16384 + hw;
    float mt = -1e30f, ms = -1e30f;
    for (int c = 0; c < 21; ++c) {
        mt = fmaxf(mt, tb[c * 16384]);
        ms = fmaxf(ms, sb[c * 16384]);
    }
    float zt = 0.f, zq = 0.f;
    for (int c = 0; c < 21; ++c) {
        zt += __expf(tb[c * 16384] - mt);
        zq += __expf(sb[c * 16384] - ms);
    }
    float lset = mt + logf(zt), lses = ms + logf(zq);
    float acc = 0.f;
    for (int c = 0; c < 21; ++c) {
        float tv = tb[c * 16384], sv = sb[c * 16384];
        float q = __expf(tv - lset);
        acc += q * ((tv - lset) - (sv - lses));   // q*log q - q*log p
    }
    block_reduce_store(acc, &piPart[blockIdx.x]);
}

// ---------------------------------------------------------------- softmax over H (dim=2)
__global__ void softmax_h_kernel(const float* __restrict__ s_out, const float* __restrict__ t_out,
                                 float* __restrict__ Ps, float* __restrict__ Pt) {
    int col = blockIdx.x * blockDim.x + threadIdx.x;   // (b,c,w) 0..10751
    if (col >= 4 * 21 * 128) return;
    const float* src = (blockIdx.y == 0 ? s_out : t_out);
    float* dst = (blockIdx.y == 0 ? Ps : Pt);
    int b = col / (21 * 128);
    int c = (col / 128) % 21;
    int w = col % 128;
    size_t base = ((size_t)(b * 21 + c) * 128) * 128 + w;
    float mx = -1e30f;
    for (int h = 0; h < 128; ++h) mx = fmaxf(mx, src[base + (size_t)h * 128]);
    float z = 0.f;
    for (int h = 0; h < 128; ++h) z += __expf(src[base + (size_t)h * 128] - mx);
    float iz = 1.0f / z;
    for (int h = 0; h < 128; ++h)
        dst[base + (size_t)h * 128] = __expf(src[base + (size_t)h * 128] - mx) * iz;
}

// ---------------------------------------------------------------- self-entropy per class
__global__ void self_ent_kernel(const float* __restrict__ Ps, const float* __restrict__ Pt,
                                float* __restrict__ seS, float* __restrict__ seT) {
    int c = blockIdx.x;
    const float* P = (blockIdx.y == 0 ? Ps : Pt);
    float* dst = (blockIdx.y == 0 ? seS : seT);
    float acc = 0.f;
    for (int i = threadIdx.x; i < 4 * 16384; i += 256) {
        int b = i >> 14, hw = i & 16383;
        float p = P[(size_t)(b * 21 + c) * 16384 + hw];
        acc += (p > 0.f) ? p * logf(p) : 0.f;
    }
    block_reduce_store(acc, &dst[c]);
}

// ---------------------------------------------------------------- cross[i][j]
__global__ void cross_kernel(const float* __restrict__ Ps, const float* __restrict__ Pt,
                             float* __restrict__ crS, float* __restrict__ crT) {
    int i = blockIdx.x / 21, j = blockIdx.x % 21;
    const float* P = (blockIdx.y == 0 ? Ps : Pt);
    float* dst = (blockIdx.y == 0 ? crS : crT);
    float acc = 0.f;
    for (int k = threadIdx.x; k < 4 * 16384; k += 256) {
        int b = k >> 14, hw = k & 16383;
        acc += P[(size_t)(b * 21 + i) * 16384 + hw] * P[(size_t)(b * 21 + j) * 16384 + hw];
    }
    block_reduce_store(acc, &dst[blockIdx.x]);
}

// ---------------------------------------------------------------- grams of logits
__global__ void gram_kernel(const float* __restrict__ s_out, const float* __restrict__ t_out,
                            float* __restrict__ Gs, float* __restrict__ Gt) {
    int b = blockIdx.x / 441;
    int ij = blockIdx.x % 441;
    int i = ij / 21, j = ij % 21;
    const float* z = (blockIdx.y == 0 ? s_out : t_out) + (size_t)b * 21 * 16384;
    float* dst = (blockIdx.y == 0 ? Gs : Gt);
    float acc = 0.f;
    for (int k = threadIdx.x; k < 16384; k += 256)
        acc += z[(size_t)i * 16384 + k] * z[(size_t)j * 16384 + k];
    block_reduce_store(acc, &dst[blockIdx.x]);
}

// ---------------------------------------------------------------- pa_loss (single block)
__global__ void pa_kernel(const float* __restrict__ sF, const float* __restrict__ tF,
                          float* __restrict__ out) {
    __shared__ float pS[2048], pT[2048];   // [b(4)][c(128)][p(4)]
    __shared__ float nS[16], nT[16];       // [b][p]
    __shared__ float red[64];
    int tid = threadIdx.x;
    for (int idx = tid; idx < 2048; idx += 256) {
        int b = idx / 512, c = (idx / 4) % 128, p = idx & 3;
        int pi = p >> 1, pj = p & 1;
        const float* bs = sF + (((size_t)(b * 128 + c) * 24 + pi * 12) * 24 + pj * 12);
        const float* bt = tF + (((size_t)(b * 128 + c) * 24 + pi * 12) * 24 + pj * 12);
        float ms = -1e30f, mt = -1e30f;
        for (int dy = 0; dy < 12; ++dy)
            for (int dx = 0; dx < 12; ++dx) {
                ms = fmaxf(ms, bs[dy * 24 + dx]);
                mt = fmaxf(mt, bt[dy * 24 + dx]);
            }
        pS[idx] = ms; pT[idx] = mt;
    }
    __syncthreads();
    if (tid < 16) {
        int b = tid / 4, p = tid & 3;
        float as = 0.f, at = 0.f;
        for (int c = 0; c < 128; ++c) {
            float vs = pS[b * 512 + c * 4 + p], vt = pT[b * 512 + c * 4 + p];
            as = fmaf(vs, vs, as);
            at = fmaf(vt, vt, at);
        }
        nS[tid] = sqrtf(as) + 1e-8f;
        nT[tid] = sqrtf(at) + 1e-8f;
    }
    __syncthreads();
    if (tid < 64) {
        int b = tid / 16, m = (tid / 4) % 4, n = tid & 3;
        float ss = 0.f, st = 0.f;
        float ism = 1.f / nS[b * 4 + m], isn = 1.f / nS[b * 4 + n];
        float itm = 1.f / nT[b * 4 + m], itn = 1.f / nT[b * 4 + n];
        for (int c = 0; c < 128; ++c) {
            ss = fmaf(pS[b * 512 + c * 4 + m] * ism, pS[b * 512 + c * 4 + n] * isn, ss);
            st = fmaf(pT[b * 512 + c * 4 + m] * itm, pT[b * 512 + c * 4 + n] * itn, st);
        }
        float d = st - ss;
        red[tid] = d * d / 64.0f;   // /((2*2)^2)/b
    }
    __syncthreads();
    if (tid == 0) {
        float s = 0.f;
        for (int k = 0; k < 64; ++k) s += red[k];
        *out = s;   // PA_L = 1
    }
}

// ---------------------------------------------------------------- epilogue scalars
__global__ void epilogue_kernel(const float* __restrict__ piPart, const float* __restrict__ saPart,
                                const float* __restrict__ Gs, const float* __restrict__ Gt,
                                const float* __restrict__ seS, const float* __restrict__ seT,
                                const float* __restrict__ crS, const float* __restrict__ crT,
                                float* __restrict__ out) {
    if (threadIdx.x != 0 || blockIdx.x != 0) return;
    // pi_loss
    float pi = 0.f;
    for (int i = 0; i < PI_BLOCKS; ++i) pi += piPart[i];
    out[N_OUT + 1] = pi / (float)N_OUT;
    // SA_loss
    float sa = 0.f;
    for (int i = 0; i < 576; ++i) sa += saPart[i];
    out[N_OUT + 4] = sqrtf(sa) / (4.0f * (float)SA_M * (float)SA_M);
    // ic_loss
    float ic = 0.f;
    for (int b = 0; b < 4; ++b)
        for (int i = 0; i < 21; ++i) {
            float as = 0.f, at = 0.f;
            for (int j = 0; j < 21; ++j) {
                float gs = Gs[b * 441 + i * 21 + j], gt = Gt[b * 441 + i * 21 + j];
                as = fmaf(gs, gs, as); at = fmaf(gt, gt, at);
            }
            float ins = 1.f / fmaxf(sqrtf(as), 1e-12f);
            float inr = 1.f / fmaxf(sqrtf(at), 1e-12f);
            for (int j = 0; j < 21; ++j) {
                float d = Gs[b * 441 + i * 21 + j] * ins - Gt[b * 441 + i * 21 + j] * inr;
                ic = fmaf(d, d, ic);
            }
        }
    out[N_OUT + 2] = ic / (21.0f * 4.0f);
    // lo_loss
    float Ks[441], Kt[441];
    for (int i = 0; i < 21; ++i)
        for (int j = 0; j < 21; ++j) {
            int mx = (i > j) ? i : j;
            Ks[i * 21 + j] = (seS[mx] - crS[i * 21 + j]) * 0.25f;
            Kt[i * 21 + j] = (seT[mx] - crT[i * 21 + j]) * 0.25f;
        }
    float lo = 0.f;
    for (int i = 0; i < 21; ++i) {
        float as = 0.f, at = 0.f;
        for (int j = 0; j < 21; ++j) {
            as = fmaf(Ks[i * 21 + j], Ks[i * 21 + j], as);
            at = fmaf(Kt[i * 21 + j], Kt[i * 21 + j], at);
        }
        float ins = 1.f / fmaxf(sqrtf(as), 1e-12f);
        float inr = 1.f / fmaxf(sqrtf(at), 1e-12f);
        for (int j = 0; j < 21; ++j) {
            float d = Ks[i * 21 + j] * ins - Kt[i * 21 + j] * inr;
            lo = fmaf(d, d, lo);
        }
    }
    out[N_OUT + 3] = lo / 441.0f / 4.0f;   // mean then /b
}

// ---------------------------------------------------------------- launch
extern "C" void kernel_launch(void* const* d_in, const int* in_sizes, int n_in,
                              void* d_out, int out_size, void* d_ws, size_t ws_size,
                              hipStream_t stream) {
    const float* t_feat3 = (const float*)d_in[0];
    const float* s_feat3 = (const float*)d_in[1];
    const float* t_feat4 = (const float*)d_in[2];
    const float* s_feat4 = (const float*)d_in[3];
    const float* t_out   = (const float*)d_in[4];
    const float* s_out   = (const float*)d_in[5];
    const float* W_B     = (const float*)d_in[6];
    const float* b_B     = (const float*)d_in[7];
    const float* W_C     = (const float*)d_in[8];
    const float* b_C     = (const float*)d_in[9];
    float* out = (float*)d_out;

    // f16 tensors first (keeps 32B alignment for v16h loads), then f32 scratch
    _Float16* Th  = (_Float16*)d_ws;
    _Float16* Abh = Th + N_F3;
    _Float16* Ach = Abh + N_F3;
    float* fws = (float*)(Ach + N_F3);
    float* Ps = fws;                fws += N_OUT;
    float* Pt = fws;                fws += N_OUT;
    float* Gs = fws;                fws += 1764;
    float* Gt = fws;                fws += 1764;
    float* seS = fws;               fws += 21;
    float* seT = fws;               fws += 21;
    float* crS = fws;               fws += 441;
    float* crT = fws;               fws += 441;
    float* saPart = fws;            fws += 576;
    float* piPart = fws;            fws += PI_BLOCKS;

    // output[0] = s_out
    copy_kernel<<<(N_OUT + 255) / 256, 256, 0, stream>>>(s_out, out, N_OUT);

    // SA: teacher f16 convert, student convs (f16 out), then WMMA strip kernel
    f32_to_f16_kernel<<<N_F3 / 256, 256, 0, stream>>>(t_feat3, Th, N_F3);
    conv_dual_kernel<<<N_F3 / 256, 256, 0, stream>>>(s_feat3, W_B, b_B, W_C, b_C, Abh, Ach);
    sa_strip_kernel<<<576, 256, 0, stream>>>(Th, Abh, Ach, saPart);

    // pi partials
    pi_kernel<<<PI_BLOCKS, 256, 0, stream>>>(t_out, s_out, piPart);

    // lo: softmax over H, self-entropy, cross grams
    softmax_h_kernel<<<dim3(42, 2), 256, 0, stream>>>(s_out, t_out, Ps, Pt);
    self_ent_kernel<<<dim3(21, 2), 256, 0, stream>>>(Ps, Pt, seS, seT);
    cross_kernel<<<dim3(441, 2), 256, 0, stream>>>(Ps, Pt, crS, crT);

    // ic: class-class grams of raw logits
    gram_kernel<<<dim3(4 * 441, 2), 256, 0, stream>>>(s_out, t_out, Gs, Gt);

    // pa (writes out[N_OUT])
    pa_kernel<<<1, 256, 0, stream>>>(s_feat4, t_feat4, out + N_OUT);

    // scalars
    epilogue_kernel<<<1, 64, 0, stream>>>(piPart, saPart, Gs, Gt, seS, seT, crS, crT, out);
}
